// FastRerank_59949153517765
// MI455X (gfx1250) — matrix operand order repaired and checked by reading, not compiled, not run on hardware
//
#include <hip/hip_runtime.h>
#include <hip/hip_bf16.h>

typedef __attribute__((ext_vector_type(16))) _Float16 v16h;
typedef __attribute__((ext_vector_type(8)))  _Float16 v8h;
typedef __attribute__((ext_vector_type(4)))  _Float16 v4h;
typedef __attribute__((ext_vector_type(8)))  float    v8f;

// ---------------------------------------------------------------------------
// WMMA helpers (CDNA5 gfx1250, wave32). D = A(16x32 f16) * B(32x16 f16) + C(f32)
// ---------------------------------------------------------------------------
__device__ __forceinline__ v8f wmma_f16(v16h a, v16h b, v8f c) {
    return __builtin_amdgcn_wmma_f32_16x16x32_f16(
        /*neg_a=*/false, a, /*neg_b=*/false, b,
        /*c_mod=*/(short)0, c, /*reuse_a=*/false, /*reuse_b=*/false);
}

// A fragment: row-major matrix, row stride ld (halves). base points at (row0,k0).
// ISA layout (16-bit A 16x32): lane&15 = M, lanes>=16 take K halves 8..15 / 24..31.
__device__ __forceinline__ v16h load_a_frag(const _Float16* base, int ld, int lane) {
    const int row = lane & 15;
    const int kb  = (lane >> 4) * 8;
    union { v16h v; v8h h[2]; } u;
    u.h[0] = *(const v8h*)(base + (size_t)row * ld + kb);
    u.h[1] = *(const v8h*)(base + (size_t)row * ld + kb + 16);
    return u.v;
}

// B fragment: storage is [n][k] (each output column has contiguous K), stride ld.
// ISA layout (16-bit B 32x16): lane&15 = N, lanes>=16 take K halves 8..15 / 24..31 (matches A).
__device__ __forceinline__ v16h load_b_frag(const _Float16* base, int ld, int lane) {
    const int n  = lane & 15;
    const int kb = (lane >> 4) * 8;
    union { v16h v; v8h h[2]; } u;
    u.h[0] = *(const v8h*)(base + (size_t)n * ld + kb);
    u.h[1] = *(const v8h*)(base + (size_t)n * ld + kb + 16);
    return u.v;
}

// ---------------------------------------------------------------------------
// 1) Convert + transpose conv weights: src [blk][tap][D=256][2D=512] fp32
//    -> dst [blk][tap][N=512][K=256] f16
// ---------------------------------------------------------------------------
__global__ __launch_bounds__(256)
void k_wconv(const float* __restrict__ exp_w, const float* __restrict__ ref_w,
             _Float16* __restrict__ wT) {
    const size_t total = 10ull * 3 * 512 * 256;
    size_t i = (size_t)blockIdx.x * blockDim.x + threadIdx.x;
    if (i >= total) return;
    const int d = (int)(i & 255);
    size_t t = i >> 8;
    const int n = (int)(t & 511); t >>= 9;
    const int tap = (int)(t % 3);
    const int blk = (int)(t / 3);
    const float* src = (blk < 7)
        ? exp_w + (((size_t)blk * 3 + tap) * 256 + d) * 512 + n
        : ref_w + (((size_t)(blk - 7) * 3 + tap) * 256 + d) * 512 + n;
    wT[i] = (_Float16)(*src);
}

// ---------------------------------------------------------------------------
// 2) Embedding gather -> fp32 activations + f16 copy. One float4 per thread.
// ---------------------------------------------------------------------------
__global__ __launch_bounds__(256)
void k_embed(const int* __restrict__ words, const float* __restrict__ emb,
             float* __restrict__ xf, _Float16* __restrict__ xh, int nrows) {
    size_t i = (size_t)blockIdx.x * blockDim.x + threadIdx.x;
    const size_t total = (size_t)nrows * 64;          // nrows * 256 / 4
    if (i >= total) return;
    const size_t row = i >> 6;
    const int    cw  = (int)(i & 63) * 4;
    const int    w   = words[row];
    const float4 v = *(const float4*)(emb + (size_t)w * 256 + cw);
    *(float4*)(xf + row * 256 + cw) = v;
    v4h h;
    h[0] = (_Float16)v.x; h[1] = (_Float16)v.y;
    h[2] = (_Float16)v.z; h[3] = (_Float16)v.w;
    *(v4h*)(xh + row * 256 + cw) = h;
}

// ---------------------------------------------------------------------------
// 3) Dilated Conv1d(K=3, D=256 -> 512) + bias + GLU + residual, via WMMA.
//    512 threads = 16 waves. Block: 64 output rows (one sequence).
//    Wave w owns GLU pair: a-cols [16w,16w+16) / g-cols [256+16w,...), and loops
//    4 M-subtiles so each weight B-fragment feeds 4 WMMAs (4x less L2 traffic).
// ---------------------------------------------------------------------------
__global__ __launch_bounds__(512)
void k_conv_glu(const float* __restrict__ xin, const _Float16* __restrict__ xinh,
                const _Float16* __restrict__ wT,   // [3][512][256] f16 (N-major)
                const float* __restrict__ bias,    // [512]
                float* __restrict__ xout, _Float16* __restrict__ xouth,
                int L, int dil) {
    __shared__ _Float16 As[3][64][256];            // 96 KB: 3 shifted A tiles
    const int tid  = threadIdx.x;
    const int lane = tid & 31;
    const int wv   = tid >> 5;                     // 0..15
    const int m0   = blockIdx.x * 64;              // global row base
    const int b    = m0 / L;
    const int l0   = m0 % L;

    // Stage shifted A tiles into LDS with zero padding at sequence edges.
    // 3 taps * 64 rows * 32 chunks(v8h) = 6144 chunks; 12 per thread.
    for (int c = tid; c < 6144; c += 512) {
        const int tap = c / 2048;
        const int rem = c % 2048;
        const int r   = rem >> 5;
        const int ko  = (rem & 31) * 8;
        const int l   = l0 + r + (tap - 1) * dil;
        v8h val;
#pragma unroll
        for (int q = 0; q < 8; ++q) val[q] = (_Float16)0.f;
        if (l >= 0 && l < L)
            val = *(const v8h*)(xinh + ((size_t)b * L + l) * 256 + ko);
        *(v8h*)(&As[tap][r][ko]) = val;
    }
    __syncthreads();

    const int na = wv * 16;            // a-column base (0..255)
    const int ng = 256 + na;           // g-column base
    v8f acc_a[4] = {};
    v8f acc_g[4] = {};
#pragma unroll
    for (int tap = 0; tap < 3; ++tap) {
        const _Float16* wa = wT + ((size_t)tap * 512 + na) * 256;
        const _Float16* wg = wT + ((size_t)tap * 512 + ng) * 256;
        __builtin_prefetch(wa, 0, 1);
        __builtin_prefetch(wg, 0, 1);
#pragma unroll
        for (int kk = 0; kk < 256; kk += 32) {
            const v16h bfa = load_b_frag(wa + kk, 256, lane);
            const v16h bfg = load_b_frag(wg + kk, 256, lane);
#pragma unroll
            for (int mt = 0; mt < 4; ++mt) {
                const v16h af = load_a_frag(&As[tap][mt * 16][kk], 256, lane);
                acc_a[mt] = wmma_f16(af, bfa, acc_a[mt]);
                acc_g[mt] = wmma_f16(af, bfg, acc_g[mt]);
            }
        }
    }

    // Epilogue: bias + GLU + residual. C/D layout: row = r + 8*(lane>>4), col = lane&15.
    const int col   = lane & 15;
    const int rbase = (lane >> 4) * 8;
    const float ba = bias[na + col];
    const float bg = bias[ng + col];
#pragma unroll
    for (int mt = 0; mt < 4; ++mt) {
#pragma unroll
        for (int r = 0; r < 8; ++r) {
            const int   row = mt * 16 + rbase + r;
            const size_t idx = ((size_t)m0 + row) * 256 + (na + col);
            const float a = acc_a[mt][r] + ba;
            const float g = acc_g[mt][r] + bg;
            const float s = 1.f / (1.f + __expf(-g));
            const float o = xin[idx] + a * s;
            xout[idx]  = o;
            xouth[idx] = (_Float16)o;
        }
    }
}

// ---------------------------------------------------------------------------
// 4) Row squared norms: one wave per row of [nrows, 256].
// ---------------------------------------------------------------------------
__global__ __launch_bounds__(256)
void k_rownorm(const float* __restrict__ x, float* __restrict__ out, int nrows) {
    const int row  = blockIdx.x * 8 + (threadIdx.x >> 5);
    const int lane = threadIdx.x & 31;
    if (row >= nrows) return;
    const float* p = x + (size_t)row * 256;
    float s = 0.f;
    for (int i = lane; i < 256; i += 32) s += p[i] * p[i];
#pragma unroll
    for (int off = 16; off > 0; off >>= 1) s += __shfl_xor(s, off);
    if (lane == 0) out[row] = s;
}

// ---------------------------------------------------------------------------
// 5) Batched dot [A,256]x[256,T] via WMMA, fused exp(-dist)*masks and row-max.
//    Block = (b, 64-row A tile); 8 waves sweep the 32 T tiles; each template
//    B-fragment feeds 4 M-subtile WMMAs.
// ---------------------------------------------------------------------------
__global__ __launch_bounds__(256)
void k_dot_max(const _Float16* __restrict__ arth, const _Float16* __restrict__ tplh,
               const float* __restrict__ a2, const float* __restrict__ t2,
               const float* __restrict__ amask, const float* __restrict__ tmask,
               float* __restrict__ rowmax) {
    __shared__ _Float16 Ash[64][256];              // 32 KB A tile
    __shared__ float    red[64][128];              // 32 KB reduction
    const int tid  = threadIdx.x;
    const int lane = tid & 31;
    const int wv   = tid >> 5;                     // 0..7
    const int b    = blockIdx.x >> 4;              // A/64 = 16 tiles per batch
    const int a0   = (blockIdx.x & 15) * 64;
    const _Float16* A = arth + ((size_t)b * 1024 + a0) * 256;

    for (int c = tid; c < 2048; c += 256) {        // 64 rows * 32 v8h chunks
        const int r = c >> 5, ko = (c & 31) * 8;
        *(v8h*)&Ash[r][ko] = *(const v8h*)(A + (size_t)r * 256 + ko);
    }
    __syncthreads();

    const int col   = lane & 15;
    const int rbase = (lane >> 4) * 8;
    float mv[4][8];
#pragma unroll
    for (int mt = 0; mt < 4; ++mt)
#pragma unroll
        for (int r = 0; r < 8; ++r) mv[mt][r] = -1e30f;

    for (int tt = wv; tt < 32; tt += 8) {
        const int t0 = tt * 16;
        const _Float16* Bp = tplh + ((size_t)b * 512 + t0) * 256;
        v8f acc[4] = {};
#pragma unroll
        for (int kk = 0; kk < 256; kk += 32) {
            const v16h bf = load_b_frag(Bp + kk, 256, lane);
#pragma unroll
            for (int mt = 0; mt < 4; ++mt) {
                const v16h af = load_a_frag(&Ash[mt * 16][kk], 256, lane);
                acc[mt] = wmma_f16(af, bf, acc[mt]);
            }
        }
        const float tn = t2[b * 512 + t0 + col];
        const float tm = tmask[b * 512 + t0 + col];
#pragma unroll
        for (int mt = 0; mt < 4; ++mt) {
#pragma unroll
            for (int r = 0; r < 8; ++r) {
                const int row = mt * 16 + rbase + r;
                const float an = a2[b * 1024 + a0 + row];
                const float am = amask[b * 1024 + a0 + row];
                float dist = fmaxf(an + tn - 2.f * acc[mt][r], 0.f);
                const float S = __expf(-dist) * am * tm;
                mv[mt][r] = fmaxf(mv[mt][r], S);
            }
        }
    }
#pragma unroll
    for (int mt = 0; mt < 4; ++mt)
#pragma unroll
        for (int r = 0; r < 8; ++r)
            red[mt * 16 + rbase + r][wv * 16 + col] = mv[mt][r];
    __syncthreads();
    if (tid < 64) {
        float m = -1e30f;
        for (int j = 0; j < 128; ++j) m = fmaxf(m, red[tid][j]);
        rowmax[(size_t)b * 1024 + a0 + tid] = m;
    }
}

// ---------------------------------------------------------------------------
// 6) Per-batch top-10 (sorted desc) + 10->10 ReLU -> 1 MLP.
// ---------------------------------------------------------------------------
__global__ __launch_bounds__(256)
void k_topff(const float* __restrict__ rowmax,
             const float* __restrict__ ff1w, const float* __restrict__ ff1b,
             const float* __restrict__ ff2w, const float* __restrict__ ff2b,
             float* __restrict__ out) {
    __shared__ float vals[1024];
    __shared__ float rv[256];
    __shared__ int   ri[256];
    __shared__ float top[10];
    const int b = blockIdx.x, tid = threadIdx.x;
    for (int i = tid; i < 1024; i += 256) vals[i] = rowmax[(size_t)b * 1024 + i];
    __syncthreads();
    for (int it = 0; it < 10; ++it) {
        float m = -1e30f; int mi = 0;
        for (int i = tid; i < 1024; i += 256)
            if (vals[i] > m) { m = vals[i]; mi = i; }
        rv[tid] = m; ri[tid] = mi;
        __syncthreads();
        for (int s = 128; s > 0; s >>= 1) {
            if (tid < s && rv[tid + s] > rv[tid]) { rv[tid] = rv[tid + s]; ri[tid] = ri[tid + s]; }
            __syncthreads();
        }
        if (tid == 0) { top[it] = rv[0]; vals[ri[0]] = -1e30f; }
        __syncthreads();
    }
    if (tid == 0) {
        float o = ff2b[0];
        for (int j = 0; j < 10; ++j) {
            float h = ff1b[j];
            for (int i = 0; i < 10; ++i) h += top[i] * ff1w[i * 10 + j];
            o += fmaxf(h, 0.f) * ff2w[j];
        }
        out[b] = o;
    }
}

// ---------------------------------------------------------------------------
// Host launcher
// ---------------------------------------------------------------------------
extern "C" void kernel_launch(void* const* d_in, const int* in_sizes, int n_in,
                              void* d_out, int out_size, void* d_ws, size_t ws_size,
                              hipStream_t stream) {
    (void)in_sizes; (void)n_in; (void)out_size; (void)ws_size;
    const int*   aw    = (const int*)d_in[0];
    const int*   tw    = (const int*)d_in[2];
    const float* amask = (const float*)d_in[4];
    const float* tmask = (const float*)d_in[5];
    const float* emb   = (const float*)d_in[6];
    const float* exp_w = (const float*)d_in[7];
    const float* exp_b = (const float*)d_in[8];
    const float* ref_w = (const float*)d_in[9];
    const float* ref_b = (const float*)d_in[10];
    const float* ff1w  = (const float*)d_in[11];
    const float* ff1b  = (const float*)d_in[12];
    const float* ff2w  = (const float*)d_in[13];
    const float* ff2b  = (const float*)d_in[14];
    float* out = (float*)d_out;

    constexpr size_t B = 16, A = 1024, T = 512, D = 256;
    constexpr size_t SZ_ARTF = B * A * D * sizeof(float);
    constexpr size_t SZ_TPLF = B * T * D * sizeof(float);
    constexpr size_t SZ_ARTH = B * A * D * sizeof(_Float16);
    constexpr size_t SZ_TPLH = B * T * D * sizeof(_Float16);
    constexpr size_t SZ_WT   = 10ull * 3 * 512 * 256 * sizeof(_Float16);

    char* ws = (char*)d_ws;
    size_t off = 0;
    auto alloc = [&](size_t bytes) { size_t o = off; off += (bytes + 255) & ~(size_t)255; return o; };
    float*    artF[2] = { (float*)(ws + alloc(SZ_ARTF)), (float*)(ws + alloc(SZ_ARTF)) };
    float*    tplF[2] = { (float*)(ws + alloc(SZ_TPLF)), (float*)(ws + alloc(SZ_TPLF)) };
    _Float16* artH[2] = { (_Float16*)(ws + alloc(SZ_ARTH)), (_Float16*)(ws + alloc(SZ_ARTH)) };
    _Float16* tplH[2] = { (_Float16*)(ws + alloc(SZ_TPLH)), (_Float16*)(ws + alloc(SZ_TPLH)) };
    _Float16* wT      = (_Float16*)(ws + alloc(SZ_WT));
    float*    a2      = (float*)(ws + alloc(B * A * sizeof(float)));
    float*    t2      = (float*)(ws + alloc(B * T * sizeof(float)));
    float*    rmax    = (float*)(ws + alloc(B * A * sizeof(float)));

    // 1) weights -> f16 transposed
    {
        const size_t total = 10ull * 3 * 512 * 256;
        k_wconv<<<dim3((unsigned)((total + 255) / 256)), dim3(256), 0, stream>>>(exp_w, ref_w, wT);
    }
    // 2) embedding gathers
    k_embed<<<dim3((unsigned)(B * A * 64 / 256)), dim3(256), 0, stream>>>(aw, emb, artF[0], artH[0], (int)(B * A));
    k_embed<<<dim3((unsigned)(B * T * 64 / 256)), dim3(256), 0, stream>>>(tw, emb, tplF[0], tplH[0], (int)(B * T));

    // 3) 10 residual conv-GLU blocks (7 expand + 3 refine), ping-pong buffers
    const int dils[10] = {1, 2, 4, 8, 16, 32, 32, 1, 1, 1};
    int ca = 0, ct = 0;
    for (int blk = 0; blk < 10; ++blk) {
        const _Float16* w    = wT + (size_t)blk * 3 * 512 * 256;
        const float*    bias = (blk < 7) ? exp_b + (size_t)blk * 512
                                         : ref_b + (size_t)(blk - 7) * 512;
        k_conv_glu<<<dim3((unsigned)(B * A / 64)), dim3(512), 0, stream>>>(
            artF[ca], artH[ca], w, bias, artF[1 - ca], artH[1 - ca], (int)A, dils[blk]);
        ca ^= 1;
        k_conv_glu<<<dim3((unsigned)(B * T / 64)), dim3(512), 0, stream>>>(
            tplF[ct], tplH[ct], w, bias, tplF[1 - ct], tplH[1 - ct], (int)T, dils[blk]);
        ct ^= 1;
    }

    // 4) row norms
    k_rownorm<<<dim3((unsigned)(B * A / 8)), dim3(256), 0, stream>>>(artF[ca], a2, (int)(B * A));
    k_rownorm<<<dim3((unsigned)(B * T / 8)), dim3(256), 0, stream>>>(tplF[ct], t2, (int)(B * T));

    // 5) pairwise exp(-dist) + row max (64-row A tiles)
    k_dot_max<<<dim3((unsigned)(B * (A / 64))), dim3(256), 0, stream>>>(
        artH[ca], tplH[ct], a2, t2, amask, tmask, rmax);

    // 6) top-10 + MLP
    k_topff<<<dim3((unsigned)B), dim3(256), 0, stream>>>(rmax, ff1w, ff1b, ff2w, ff2b, out);
}
